// HungarianMatcher_34170759807565
// MI455X (gfx1250) — matrix operand order repaired and checked by reading, not compiled, not run on hardware
//
#include <hip/hip_runtime.h>
#include <hip/hip_bf16.h>

// ---- problem constants (match reference setup_inputs) ----
#define B_DIM 16
#define N_DIM 900
#define T_DIM 256
#define M_DIM 100
#define BN (B_DIM * N_DIM)   // 14400
#define BM (B_DIM * M_DIM)   // 1600

#define F_ALPHA 0.25f
#define F_EPS   1e-8f
#define W_CLASS 1.0f
#define W_BBOX  5.0f
#define W_GIOU  2.0f

typedef __attribute__((ext_vector_type(16))) _Float16 v16h;
typedef __attribute__((ext_vector_type(8)))  _Float16 v8h;
typedef __attribute__((ext_vector_type(8)))  float    v8f;

// ---------------------------------------------------------------------------
// Kernel 1: per (row, t): prob = sigmoid(logit)
//   neg = (1-a)*p^2*(-log(1-p+eps)); pos = a*(1-p)^2*(-log(p+eps))
//   combined[row,t] = f16(pos+neg);  rowsum_neg[row] = sum_t neg
// one 256-thread block per row (T == 256)
// ---------------------------------------------------------------------------
__global__ __launch_bounds__(256) void matcher_prep_logits(
    const float* __restrict__ logits,
    _Float16* __restrict__ combined,
    float* __restrict__ rowsum_neg)
{
    __shared__ float red[256];
    const int row = blockIdx.x;
    const int t   = threadIdx.x;
    const float x = logits[row * T_DIM + t];
    const float p = 1.0f / (1.0f + __expf(-x));
    const float q = 1.0f - p;
    const float neg = (1.0f - F_ALPHA) * p * p * (-__logf(q + F_EPS));
    const float pos = F_ALPHA * q * q * (-__logf(p + F_EPS));
    combined[row * T_DIM + t] = (_Float16)(pos + neg);
    red[t] = neg;
    __syncthreads();
#pragma unroll
    for (int s = 128; s > 0; s >>= 1) {
        if (t < s) red[t] += red[t + s];
        __syncthreads();
    }
    if (t == 0) rowsum_neg[row] = red[0];
}

// ---------------------------------------------------------------------------
// Kernel 2: cast binary token labels f32 -> f16 (exact)
// ---------------------------------------------------------------------------
__global__ __launch_bounds__(256) void matcher_prep_labels(
    const float* __restrict__ tl, _Float16* __restrict__ tlh, int n)
{
    const int i = blockIdx.x * 256 + threadIdx.x;
    if (i < n) tlh[i] = (_Float16)tl[i];
}

// ---------------------------------------------------------------------------
// Kernel 3: fused WMMA GEMM + box-cost epilogue.
// Each wave: one 16-row tile x 4 col-tiles (64 cols). K = 256 in steps of 32.
// Fragment layouts loaded directly from row-major f16 (same contiguous k order for A and B):
//   A 16x32: lane -> row=lane&15, 16 contiguous halfs at k + 16*(lane>>4)
//   B 32x16: lane -> col=lane&15, 16 contiguous halfs at k + 16*(lane>>4)
//   C/D f32: vgpr v, lane l -> M = v + 8*(l>>4), N = l&15
// ---------------------------------------------------------------------------
__global__ __launch_bounds__(256) void matcher_wmma_fused(
    const _Float16* __restrict__ Acomb,      // [BN, T]
    const _Float16* __restrict__ Btl,        // [BM, T]
    const float*    __restrict__ rowsum_neg, // [BN]
    const float*    __restrict__ pred_boxes, // [BN, 4] cxcywh
    const float*    __restrict__ tgt_boxes,  // [BM, 4] cxcywh
    float*          __restrict__ out)        // [BN, BM]
{
    const int lane  = threadIdx.x & 31;
    const int wave  = threadIdx.x >> 5;
    const int W     = blockIdx.x * 8 + wave;
    const int COL_GROUPS = BM / 64;                 // 25
    const int TOTAL = (BN / 16) * COL_GROUPS;       // 22500 waves
    if (W >= TOTAL) return;

    const int row_tile  = W / COL_GROUPS;
    const int col_group = W % COL_GROUPS;
    const int row_base  = row_tile * 16;
    const int col_base  = col_group * 64;

    const int lhalf = lane >> 4;    // 0 or 1
    const int lmod  = lane & 15;

    // per-lane base pointer into A for this lane's fragment rows
    const _Float16* aptr = Acomb + (row_base + lmod) * T_DIM + lhalf * 16;

    v8f acc[4] = {};

    for (int k = 0; k < T_DIM; k += 32) {
        if (k + 32 < T_DIM)
            __builtin_prefetch(aptr + k + 32, 0, 1);   // global_prefetch next A slice
        v8h alo = *(const v8h*)(aptr + k);
        v8h ahi = *(const v8h*)(aptr + k + 8);
        v16h a = __builtin_shufflevector(alo, ahi,
                 0, 1, 2, 3, 4, 5, 6, 7, 8, 9, 10, 11, 12, 13, 14, 15);
#pragma unroll
        for (int c = 0; c < 4; ++c) {
            const _Float16* bptr =
                Btl + (col_base + c * 16 + lmod) * T_DIM + k + lhalf * 16;
            v16h b = *(const v16h*)bptr;
            acc[c] = __builtin_amdgcn_wmma_f32_16x16x32_f16(
                false, a, false, b, (short)0, acc[c], false, false);
        }
    }

    // ---- epilogue: this lane owns rows row_base + lhalf*8 + v, col = c*16+lmod
    const int m0 = lhalf * 8;
    float4 pb[8];
    float  rs[8];
#pragma unroll
    for (int v = 0; v < 8; ++v) {
        const int row = row_base + m0 + v;
        pb[v] = *(const float4*)(pred_boxes + row * 4);
        rs[v] = rowsum_neg[row];
    }

#pragma unroll
    for (int c = 0; c < 4; ++c) {
        const int col = col_base + c * 16 + lmod;
        const float4 tb = *(const float4*)(tgt_boxes + col * 4);
        const float tx0 = tb.x - 0.5f * tb.z, ty0 = tb.y - 0.5f * tb.w;
        const float tx1 = tb.x + 0.5f * tb.z, ty1 = tb.y + 0.5f * tb.w;
        const float tarea = tb.z * tb.w;
#pragma unroll
        for (int v = 0; v < 8; ++v) {
            const int row = row_base + m0 + v;
            const float4 p = pb[v];
            // L1 on raw cxcywh
            const float cost_bbox = fabsf(p.x - tb.x) + fabsf(p.y - tb.y)
                                  + fabsf(p.z - tb.z) + fabsf(p.w - tb.w);
            // GIoU on xyxy
            const float px0 = p.x - 0.5f * p.z, py0 = p.y - 0.5f * p.w;
            const float px1 = p.x + 0.5f * p.z, py1 = p.y + 0.5f * p.w;
            const float parea = p.z * p.w;
            float iw = fminf(px1, tx1) - fmaxf(px0, tx0); iw = fmaxf(iw, 0.0f);
            float ih = fminf(py1, ty1) - fmaxf(py0, ty0); ih = fmaxf(ih, 0.0f);
            const float inter = iw * ih;
            const float uni   = parea + tarea - inter;
            const float iou   = inter / uni;
            float cw = fmaxf(px1, tx1) - fminf(px0, tx0); cw = fmaxf(cw, 0.0f);
            float ch = fmaxf(py1, ty1) - fminf(py0, ty0); ch = fmaxf(ch, 0.0f);
            const float areac = cw * ch;
            const float giou  = iou - (areac - uni) / areac;

            const float cost_class = acc[c][v] - rs[v];
            out[row * BM + col] =
                W_BBOX * cost_bbox + W_CLASS * cost_class - W_GIOU * giou;
        }
    }
}

// ---------------------------------------------------------------------------
// launch
// inputs (setup_inputs order): 0=pred_logits [B,N,T] f32, 1=pred_boxes [B,N,4],
//                              2=tgt_boxes [B,M,4], 3=tgt_token_labels [B,M,T]
// output: C [B, N, B*M] f32
// workspace: combined f16 (7,372,800 B) | tlh f16 (819,200 B) | rowsum f32
// ---------------------------------------------------------------------------
extern "C" void kernel_launch(void* const* d_in, const int* in_sizes, int n_in,
                              void* d_out, int out_size, void* d_ws, size_t ws_size,
                              hipStream_t stream) {
    const float* pred_logits = (const float*)d_in[0];
    const float* pred_boxes  = (const float*)d_in[1];
    const float* tgt_boxes   = (const float*)d_in[2];
    const float* tgt_labels  = (const float*)d_in[3];
    float* out = (float*)d_out;

    char* ws = (char*)d_ws;
    _Float16* combined = (_Float16*)(ws);                       // BN*T f16
    _Float16* tlh      = (_Float16*)(ws + (size_t)BN * T_DIM * 2);
    float*    rowsum   = (float*)(ws + (size_t)BN * T_DIM * 2
                                     + (size_t)BM * T_DIM * 2);

    // 1) focal terms + per-row neg sum
    matcher_prep_logits<<<BN, 256, 0, stream>>>(pred_logits, combined, rowsum);

    // 2) labels -> f16
    const int nlab = BM * T_DIM;
    matcher_prep_labels<<<(nlab + 255) / 256, 256, 0, stream>>>(tgt_labels, tlh, nlab);

    // 3) fused WMMA GEMM + box epilogue: 22500 waves, 8 waves per block
    const int total_waves = (BN / 16) * (BM / 64);
    const int blocks = (total_waves + 7) / 8;
    matcher_wmma_fused<<<blocks, 256, 0, stream>>>(
        combined, tlh, rowsum, pred_boxes, tgt_boxes, out);
}
